// MultiHeadedDotAttention_2027224563767
// MI455X (gfx1250) — compile-verified, matrix operand order checked by reading, not run on hardware
//
#include <hip/hip_runtime.h>
#include <hip/hip_bf16.h>

typedef __attribute__((ext_vector_type(16))) __bf16 v16bf;
typedef __attribute__((ext_vector_type(8)))  __bf16 v8bf;
typedef __attribute__((ext_vector_type(8)))  float  v8f;

#define WMMA_BF16(a, b, c) \
  __builtin_amdgcn_wmma_f32_16x16x32_bf16(false, (a), false, (b), (short)0, (c), false, false)

#define NEGV  -1000000000.0f
#define QSCALE 0.08838834764831845f   // 1/sqrt(128)

// CDNA5 async copy global -> LDS (ASYNCcnt). Fallback: plain ds store.
#ifndef USE_ASYNC_LDS
#define USE_ASYNC_LDS 1
#endif

static __device__ inline void cp_b128(__bf16* l, const __bf16* g) {
#if USE_ASYNC_LDS
  const unsigned lo = (unsigned)(size_t)l;   // low 32 bits of generic shared addr = LDS offset
  asm volatile("global_load_async_to_lds_b128 %0, %1, off" :: "v"(lo), "v"(g) : "memory");
#else
  *reinterpret_cast<v8bf*>(l) = *reinterpret_cast<const v8bf*>(g);
#endif
}

static __device__ inline void async_drain() {
#if USE_ASYNC_LDS
  asm volatile("s_wait_asynccnt 0x0" ::: "memory");
#endif
}

static __device__ inline v8f zero8() {
  v8f z;
#pragma unroll
  for (int i = 0; i < 8; ++i) z[i] = 0.f;
  return z;
}

// 16x32 bf16 fragment (A-layout; also serves as B for X@W^T since both are
// row-major-by-k). Lanes 0-15: row = lane, K = {k0..k0+7, k0+16..k0+23};
// lanes 16-31: row = lane-16, K = {k0+8..k0+15, k0+24..k0+31}.
static __device__ inline v16bf ld_frag16(const __bf16* p, int ld, int row, int k0, int lane) {
  const int half = (lane >> 4) & 1;
  const int r    = lane & 15;
  const __bf16* q = p + (size_t)(row + r) * (size_t)ld + (size_t)(k0 + half * 8);
  v8bf lo = *reinterpret_cast<const v8bf*>(q);
  v8bf hi = *reinterpret_cast<const v8bf*>(q + 16);
  return __builtin_shufflevector(lo, hi, 0, 1, 2, 3, 4, 5, 6, 7, 8, 9, 10, 11, 12, 13, 14, 15);
}

// ---------------------------------------------------------------- f32 -> bf16
__global__ __launch_bounds__(256) void cvt_f32_bf16_kernel(
    const float* __restrict__ s, __bf16* __restrict__ d, int n) {
  for (int i = blockIdx.x * blockDim.x + threadIdx.x; i < n; i += gridDim.x * blockDim.x)
    d[i] = (__bf16)s[i];
}

// -------------------------------------------------- LayerNorm(query) -> bf16
__global__ __launch_bounds__(256) void layernorm_kernel(
    const float* __restrict__ x, const float* __restrict__ gam,
    const float* __restrict__ bet, __bf16* __restrict__ dst /* ld = 2048 */) {
  const int row = blockIdx.x, tid = threadIdx.x;
  const float* xr = x + (size_t)row * 1024;
  __shared__ float red[256];

  float s = 0.f;
  for (int i = tid; i < 1024; i += 256) s += xr[i];
  red[tid] = s;
  __syncthreads();
  for (int off = 128; off; off >>= 1) {
    if (tid < off) red[tid] += red[tid + off];
    __syncthreads();
  }
  const float mean = red[0] * (1.f / 1024.f);
  __syncthreads();

  float vs = 0.f;
  for (int i = tid; i < 1024; i += 256) {
    float dl = xr[i] - mean;
    vs += dl * dl;
  }
  red[tid] = vs;
  __syncthreads();
  for (int off = 128; off; off >>= 1) {
    if (tid < off) red[tid] += red[tid + off];
    __syncthreads();
  }
  const float inv = 1.f / (sqrtf(red[0] * (1.f / 1023.f)) + 1e-6f);

  __bf16* dr = dst + (size_t)row * 2048;
  for (int i = tid; i < 1024; i += 256)
    dr[i] = (__bf16)(gam[i] * (xr[i] - mean) * inv + bet[i]);
}

// ------------------------------------------------ C = A @ W^T + bias (staged)
// Workgroup 256 thr = 8 waves, tile 128(M) x 128(N); wave tile 32x64.
// Double-buffered LDS staging via async global->LDS copies.
// transpose_out == 0: C[m][n]  (ldc = N)
// transpose_out == 1: V^T store: C[(m & ~1023) + n][m & 1023], vectorized 8-wide
// Grid: (M/128, N/128).
__global__ __launch_bounds__(256) void gemm_qkv_staged(
    const __bf16* __restrict__ A, int lda,
    const __bf16* __restrict__ W, int ldw,
    const float* __restrict__ bias,
    __bf16* __restrict__ C, int ldc, int K, int transpose_out) {
  __shared__ __bf16 sA[2][128 * 32];
  __shared__ __bf16 sB[2][128 * 32];

  const int tid = threadIdx.x, wave = tid >> 5, lane = tid & 31;
  const int half = (lane >> 4) & 1, nl = lane & 15;
  const int mblk = blockIdx.x * 128, nblk = blockIdx.y * 128;
  const int mw = (wave >> 1) * 32, nw = (wave & 1) * 64;

  v8f acc[2][4];
#pragma unroll
  for (int i = 0; i < 2; ++i)
#pragma unroll
    for (int j = 0; j < 4; ++j) acc[i][j] = zero8();

  // stage one 128x32 bf16 tile (8 KB): 512 x 16B chunks, 2 per thread
  auto stage = [&](__bf16* dstL, const __bf16* src, int ldsrc, int rblk, int kb) {
#pragma unroll
    for (int c = tid; c < 512; c += 256) {
      const int row = c >> 2, ko = (c & 3) * 8;
      cp_b128(dstL + row * 32 + ko, src + (size_t)(rblk + row) * ldsrc + kb + ko);
    }
  };

  stage(sA[0], A, lda, mblk, 0);
  stage(sB[0], W, ldw, nblk, 0);
  int cur = 0;

  for (int kb = 0; kb < K; kb += 32) {
    async_drain();
    __syncthreads();           // current buffer fully staged, visible to all
    if (kb + 32 < K) {
      stage(sA[cur ^ 1], A, lda, mblk, kb + 32);
      stage(sB[cur ^ 1], W, ldw, nblk, kb + 32);
    }

    v16bf af[2], bfr[4];
#pragma unroll
    for (int i = 0; i < 2; ++i) af[i] = ld_frag16(sA[cur], 32, mw + i * 16, 0, lane);
#pragma unroll
    for (int j = 0; j < 4; ++j) bfr[j] = ld_frag16(sB[cur], 32, nw + j * 16, 0, lane);
#pragma unroll
    for (int i = 0; i < 2; ++i)
#pragma unroll
      for (int j = 0; j < 4; ++j)
        acc[i][j] = WMMA_BF16(af[i], bfr[j], acc[i][j]);

    __syncthreads();           // everyone done reading before buffer is reused
    cur ^= 1;
  }

  if (transpose_out) {
    // Per lane: n fixed, the 8 acc elems are 8 consecutive m -> one 16B store.
#pragma unroll
    for (int i = 0; i < 2; ++i)
#pragma unroll
      for (int j = 0; j < 4; ++j) {
        const int mbase = mblk + mw + i * 16 + 8 * half;
        const int n = nblk + nw + j * 16 + nl;
        v8bf t;
#pragma unroll
        for (int v = 0; v < 8; ++v) t[v] = (__bf16)(acc[i][j][v] + bias[n]);
        const size_t addr = (size_t)((mbase & ~1023) + n) * 1024 + (size_t)(mbase & 1023);
        *reinterpret_cast<v8bf*>(&C[addr]) = t;
      }
  } else {
#pragma unroll
    for (int i = 0; i < 2; ++i)
#pragma unroll
      for (int j = 0; j < 4; ++j)
#pragma unroll
        for (int v = 0; v < 8; ++v) {
          const int m = mblk + mw + i * 16 + v + 8 * half;
          const int n = nblk + nw + j * 16 + nl;
          C[(size_t)m * ldc + n] = (__bf16)(acc[i][j][v] + bias[n]);
        }
  }
}

// ------------------------------------------------------------------ attention
// One block (8 waves) per (b, h, 16-query tile). Flash-style over 2 chunks of
// 512 keys: scores via WMMA -> LDS, online softmax, P@V via WMMA (P from LDS,
// V fragments from the transposed Vt with vectorized b128 loads).
__global__ __launch_bounds__(256) void attn_kernel(
    const __bf16* __restrict__ Qp, const __bf16* __restrict__ Kp,
    const __bf16* __restrict__ Vt /* (B*1024) x 1024, V^T per batch */,
    const int* __restrict__ mask,
    __bf16* __restrict__ xq /* 8192 x 2048 */) {
  const int tid = threadIdx.x;
  const int wave = tid >> 5, lane = tid & 31;
  const int half = (lane >> 4) & 1, nl = lane & 15;

  const int qt = blockIdx.x & 63;
  const int h  = (blockIdx.x >> 6) & 7;
  const int b  = blockIdx.x >> 9;
  const int q0 = qt << 4;
  const int rowQ = b * 1024 + q0;

  __shared__ float  s_sc[16 * 512];
  __shared__ __bf16 s_p[16 * 512];
  __shared__ float  s_m[16], s_l[16], s_scale[16];

  v16bf qf[4];
#pragma unroll
  for (int kk = 0; kk < 4; ++kk)
    qf[kk] = ld_frag16(Qp + h * 128, 1024, rowQ, kk * 32, lane);

  if (tid < 16) { s_m[tid] = -3.0e38f; s_l[tid] = 0.f; }
  __syncthreads();

  v8f oacc = zero8();
  const __bf16* Vt_bh = Vt + (size_t)(b * 1024 + h * 128) * 1024;

  for (int c = 0; c < 2; ++c) {
    const int kbase = c * 512;

    for (int jt = wave; jt < 32; jt += 8) {
      const int jrow = b * 1024 + kbase + jt * 16;
      v8f sc = zero8();
#pragma unroll
      for (int kk = 0; kk < 4; ++kk) {
        v16bf kf = ld_frag16(Kp + h * 128, 1024, jrow, kk * 32, lane);
        sc = WMMA_BF16(qf[kk], kf, sc);
      }
#pragma unroll
      for (int v = 0; v < 8; ++v)
        s_sc[(v + 8 * half) * 512 + jt * 16 + nl] = sc[v];
    }
    __syncthreads();

    {
      const int r = tid >> 4, i = tid & 15;
      float mc = -3.0e38f;
      for (int j = i; j < 512; j += 16) {
        float s = s_sc[r * 512 + j];
        s = (mask[b * 1024 + kbase + j] != 0) ? s * QSCALE : NEGV;
        s_sc[r * 512 + j] = s;
        mc = fmaxf(mc, s);
      }
#pragma unroll
      for (int d = 1; d < 16; d <<= 1) mc = fmaxf(mc, __shfl_xor(mc, d, 32));
      const float m_old = s_m[r];
      const float m_new = fmaxf(m_old, mc);
      float lc = 0.f;
      for (int j = i; j < 512; j += 16) {
        float e = __expf(s_sc[r * 512 + j] - m_new);
        s_p[r * 512 + j] = (__bf16)e;
        lc += e;
      }
#pragma unroll
      for (int d = 1; d < 16; d <<= 1) lc += __shfl_xor(lc, d, 32);
      if (i == 0) {
        const float sf = __expf(m_old - m_new);
        s_l[r] = s_l[r] * sf + lc;
        s_m[r] = m_new;
        s_scale[r] = sf;
      }
    }
    __syncthreads();

    {
#pragma unroll
      for (int v = 0; v < 8; ++v) oacc[v] *= s_scale[v + 8 * half];
      for (int kt = 0; kt < 16; ++kt) {
        const int kloc = kt * 32;
        v16bf pf = ld_frag16(s_p, 512, 0, kloc, lane);
        // B-frag of V: row = head-dim column (this wave's 16-wide d tile),
        // k = sequence position -> plain row-major fragment from Vt.
        v16bf vf = ld_frag16(Vt_bh, 1024, wave * 16, kbase + kloc, lane);
        oacc = WMMA_BF16(pf, vf, oacc);
      }
    }
    __syncthreads();
  }

#pragma unroll
  for (int v = 0; v < 8; ++v) {
    const int rloc = v + 8 * half;
    const float o = oacc[v] / s_l[rloc];
    const size_t m = (size_t)(b * 1024 + q0 + rloc);
    xq[m * 2048 + h * 128 + wave * 16 + nl] = (__bf16)o;
  }
}

// --------------------- z = [x|q] @ Waoa^T + b; out = a * sigmoid(g)  (staged)
// Workgroup tile: 128(M) x 64(a-cols, + matching gate cols). Wave tile 32x32.
// Grid: (8192/128, 1024/64). K = 2048.
__global__ __launch_bounds__(256) void gemm_aoa_glu_staged(
    const __bf16* __restrict__ XQ, const __bf16* __restrict__ W,
    const float* __restrict__ bias, float* __restrict__ out) {
  __shared__ __bf16 sA[2][128 * 32];
  __shared__ __bf16 sBa[2][64 * 32];
  __shared__ __bf16 sBg[2][64 * 32];

  const int tid = threadIdx.x, wave = tid >> 5, lane = tid & 31;
  const int half = (lane >> 4) & 1, nl = lane & 15;
  const int mblk = blockIdx.x * 128, nblk = blockIdx.y * 64;
  const int mw = (wave >> 1) * 32, nw = (wave & 1) * 32;

  v8f aa[2][2], ag[2][2];
#pragma unroll
  for (int i = 0; i < 2; ++i)
#pragma unroll
    for (int j = 0; j < 2; ++j) { aa[i][j] = zero8(); ag[i][j] = zero8(); }

  auto stageA = [&](int buf, int kb) {
#pragma unroll
    for (int c = tid; c < 512; c += 256) {
      const int row = c >> 2, ko = (c & 3) * 8;
      cp_b128(&sA[buf][row * 32 + ko], XQ + (size_t)(mblk + row) * 2048 + kb + ko);
    }
  };
  auto stageB = [&](int buf, int kb) {
    const int row = tid >> 2, ko = (tid & 3) * 8;
    cp_b128(&sBa[buf][row * 32 + ko], W + (size_t)(nblk + row) * 2048 + kb + ko);
    cp_b128(&sBg[buf][row * 32 + ko], W + (size_t)(1024 + nblk + row) * 2048 + kb + ko);
  };

  stageA(0, 0);
  stageB(0, 0);
  int cur = 0;

  for (int kb = 0; kb < 2048; kb += 32) {
    async_drain();
    __syncthreads();
    if (kb + 32 < 2048) {
      stageA(cur ^ 1, kb + 32);
      stageB(cur ^ 1, kb + 32);
    }

    v16bf af[2], ba[2], bg[2];
#pragma unroll
    for (int i = 0; i < 2; ++i) af[i] = ld_frag16(sA[cur], 32, mw + i * 16, 0, lane);
#pragma unroll
    for (int j = 0; j < 2; ++j) {
      ba[j] = ld_frag16(sBa[cur], 32, nw + j * 16, 0, lane);
      bg[j] = ld_frag16(sBg[cur], 32, nw + j * 16, 0, lane);
    }
#pragma unroll
    for (int i = 0; i < 2; ++i)
#pragma unroll
      for (int j = 0; j < 2; ++j) {
        aa[i][j] = WMMA_BF16(af[i], ba[j], aa[i][j]);
        ag[i][j] = WMMA_BF16(af[i], bg[j], ag[i][j]);
      }

    __syncthreads();
    cur ^= 1;
  }

#pragma unroll
  for (int i = 0; i < 2; ++i)
#pragma unroll
    for (int j = 0; j < 2; ++j)
#pragma unroll
      for (int v = 0; v < 8; ++v) {
        const int m = mblk + mw + i * 16 + v + 8 * half;
        const int n = nblk + nw + j * 16 + nl;
        const float av = aa[i][j][v] + bias[n];
        const float gv = ag[i][j][v] + bias[1024 + n];
        out[(size_t)m * 1024 + n] = av * (1.f / (1.f + __expf(-gv)));
      }
}

extern "C" void kernel_launch(void* const* d_in, const int* in_sizes, int n_in,
                              void* d_out, int out_size, void* d_ws, size_t ws_size,
                              hipStream_t stream) {
  (void)in_sizes; (void)n_in; (void)out_size; (void)ws_size;

  const float* query = (const float*)d_in[0];
  const float* value = (const float*)d_in[1];
  const float* key   = (const float*)d_in[2];
  const int*   mask  = (const int*)d_in[3];
  const float* ln_a  = (const float*)d_in[4];
  const float* ln_b  = (const float*)d_in[5];
  const float* wq    = (const float*)d_in[6];
  const float* bq    = (const float*)d_in[7];
  const float* wk    = (const float*)d_in[8];
  const float* bk    = (const float*)d_in[9];
  const float* wv    = (const float*)d_in[10];
  const float* bv    = (const float*)d_in[11];
  const float* waoa  = (const float*)d_in[12];
  const float* baoa  = (const float*)d_in[13];
  float* out = (float*)d_out;

  char* p = (char*)d_ws;
  auto take = [&](size_t elems) {
    __bf16* r = (__bf16*)p;
    p += (elems * sizeof(__bf16) + 255) & ~(size_t)255;
    return r;
  };
  __bf16* wq_bf  = take((size_t)1024 * 1024);
  __bf16* wk_bf  = take((size_t)1024 * 1024);
  __bf16* wv_bf  = take((size_t)1024 * 1024);
  __bf16* wa_bf  = take((size_t)2048 * 2048);
  __bf16* key_bf = take((size_t)8192 * 1024);
  __bf16* val_bf = take((size_t)8192 * 1024);
  __bf16* xq_bf  = take((size_t)8192 * 2048);   // [x | layernorm(q)] concat, bf16
  __bf16* Qp     = take((size_t)8192 * 1024);
  __bf16* Kp     = take((size_t)8192 * 1024);
  __bf16* Vt     = take((size_t)8192 * 1024);   // V^T per batch: (b*1024+d) x s

  const dim3 blk(256);

  cvt_f32_bf16_kernel<<<1024, blk, 0, stream>>>(wq, wq_bf, 1024 * 1024);
  cvt_f32_bf16_kernel<<<1024, blk, 0, stream>>>(wk, wk_bf, 1024 * 1024);
  cvt_f32_bf16_kernel<<<1024, blk, 0, stream>>>(wv, wv_bf, 1024 * 1024);
  cvt_f32_bf16_kernel<<<2048, blk, 0, stream>>>(waoa, wa_bf, 2048 * 2048);
  cvt_f32_bf16_kernel<<<2048, blk, 0, stream>>>(key, key_bf, 8192 * 1024);
  cvt_f32_bf16_kernel<<<2048, blk, 0, stream>>>(value, val_bf, 8192 * 1024);

  layernorm_kernel<<<8192, blk, 0, stream>>>(query, ln_a, ln_b, xq_bf + 1024);

  // Q/K/V projections (M=8192, N=1024, K=1024), staged async->LDS WMMA GEMM.
  // V is written transposed (per batch) so attention reads it with b128 frags.
  gemm_qkv_staged<<<dim3(64, 8), blk, 0, stream>>>(xq_bf + 1024, 2048, wq_bf, 1024, bq, Qp, 1024, 1024, 0);
  gemm_qkv_staged<<<dim3(64, 8), blk, 0, stream>>>(key_bf, 1024, wk_bf, 1024, bk, Kp, 1024, 1024, 0);
  gemm_qkv_staged<<<dim3(64, 8), blk, 0, stream>>>(val_bf, 1024, wv_bf, 1024, bv, Vt, 1024, 1024, 1);

  attn_kernel<<<4096, blk, 0, stream>>>(Qp, Kp, Vt, mask, xq_bf);

  gemm_aoa_glu_staged<<<dim3(64, 16), blk, 0, stream>>>(xq_bf, wa_bf, baoa, out);
}
